// ContextAwareRegionalAttentionNetwork_6863357739503
// MI455X (gfx1250) — matrix operand order repaired and checked by reading, not compiled, not run on hardware
//
#include <hip/hip_runtime.h>
#include <hip/hip_bf16.h>

// ---------------------------------------------------------------------------
// ContextAwareRegionalAttentionNetwork for MI455X (gfx1250, wave32, WMMA)
//
// Phase 1: per-(b,c) plane SAT in LDS (25.6KB plane -> 26.2KB SAT), direct
//          box sums for ROIs of that batch -> pooled (2016 x 4096) bf16.
//          features (210MB) read exactly once from HBM (~9us floor @23.3TB/s).
// Phase 2: pooled(M=2016,K=4096) x W1^T(K,64) via v_wmma_f32_16x16x32_bf16
//          (fp32 accum), fused tanh -> W2 dot -> softplus epilogue.
// ---------------------------------------------------------------------------

#define SCALE_   0.03125f
#define B_       4
#define C_       2048
#define H_       80
#define W_       80
#define R_       2000
#define K_       (2 * C_)      // 4096
#define M_PAD    2016          // 126 tiles of 16
#define NHID     64

typedef __attribute__((ext_vector_type(16))) __bf16 v16bf;
typedef __attribute__((ext_vector_type(8)))  float  v8f;

union Frag16 {
    uint4 u[2];
    v16bf v;
};

// ---------------------------------------------------------------------------
// Kernel 1: decode ROI boxes once (round half-to-even matches jnp.round/RNE).
// ---------------------------------------------------------------------------
__global__ void prep_rois_k(const float* __restrict__ rois,
                            int* __restrict__ rb,  int* __restrict__ rhs,
                            int* __restrict__ rhe, int* __restrict__ rws,
                            int* __restrict__ rwe, int* __restrict__ rempty,
                            float* __restrict__ rinv) {
    int r = blockIdx.x * blockDim.x + threadIdx.x;
    if (r >= R_) return;
    const float* p = rois + r * 5;
    int b  = (int)p[0];
    int x1 = (int)rintf(p[1] * SCALE_);
    int y1 = (int)rintf(p[2] * SCALE_);
    int x2 = (int)rintf(p[3] * SCALE_);
    int y2 = (int)rintf(p[4] * SCALE_);
    int rw = max(x2 - x1 + 1, 1);
    int rh = max(y2 - y1 + 1, 1);
    int hs = min(max(y1, 0), H_);
    int he = min(max(y1 + rh, 0), H_);
    int ws = min(max(x1, 0), W_);
    int we = min(max(x1 + rw, 0), W_);
    int area = (he - hs) * (we - ws);
    rb[r] = b; rhs[r] = hs; rhe[r] = he; rws[r] = ws; rwe[r] = we;
    rempty[r] = ((he <= hs) || (we <= ws)) ? 1 : 0;
    rinv[r] = 1.0f / (float)max(area, 1);
}

// ---------------------------------------------------------------------------
// Kernel 2: W1 -> bf16 and zero the 16 pad rows of pooled.
// ---------------------------------------------------------------------------
__global__ void prep_weights_k(const float* __restrict__ W1,
                               __hip_bfloat16* __restrict__ w1b,
                               __hip_bfloat16* __restrict__ pooled) {
    int i = blockIdx.x * blockDim.x + threadIdx.x;
    if (i < NHID * K_) w1b[i] = __hip_bfloat16(W1[i]);
    if (i < (M_PAD - R_) * K_) pooled[(size_t)R_ * K_ + i] = __hip_bfloat16(0.0f);
}

// ---------------------------------------------------------------------------
// Kernel 3: one workgroup per (b,c) plane. SAT built in LDS (81x81 padded),
// then box sums for every ROI with matching batch index.
// ---------------------------------------------------------------------------
__global__ void __launch_bounds__(256)
plane_pool_k(const float* __restrict__ feat,
             const int* __restrict__ rb,  const int* __restrict__ rhs,
             const int* __restrict__ rhe, const int* __restrict__ rws,
             const int* __restrict__ rwe, const int* __restrict__ rempty,
             const float* __restrict__ rinv,
             __hip_bfloat16* __restrict__ pooled) {
    __shared__ float sat[81 * 81];   // 26,244 B of the 320KB/WGP LDS pool

    const int plane = blockIdx.x;          // b * C + c
    const int b = plane / C_;
    const int c = plane % C_;
    const float* f = feat + (size_t)plane * (H_ * W_);
    const int tid = threadIdx.x;

    // Load plane into SAT interior (one-element zero border at row/col 0).
    for (int i = tid; i < H_ * W_; i += 256) {
        int y = i / W_, x = i - y * W_;
        sat[(y + 1) * 81 + (x + 1)] = f[i];
    }
    for (int i = tid; i < 81; i += 256) { sat[i] = 0.0f; sat[i * 81] = 0.0f; }
    __syncthreads();

    // Row-wise inclusive prefix sums (one row per thread; 80 rows).
    if (tid < 80) {
        float run = 0.0f;
        int base = (tid + 1) * 81;
        for (int x = 1; x <= 80; ++x) { run += sat[base + x]; sat[base + x] = run; }
    }
    __syncthreads();

    // Column-wise prefix sums (consecutive threads -> consecutive banks).
    if (tid < 80) {
        float run = 0.0f;
        int col = tid + 1;
        for (int y = 1; y <= 80; ++y) { run += sat[y * 81 + col]; sat[y * 81 + col] = run; }
    }
    __syncthreads();

    const float gmean = sat[80 * 81 + 80] * (1.0f / (float)(H_ * W_));

    // Box sums for all ROIs belonging to this batch image.
    for (int r = tid; r < R_; r += 256) {
        if (rb[r] != b) continue;
        int hs = rhs[r], he = rhe[r], ws = rws[r], we = rwe[r];
        float win = sat[he * 81 + we] - sat[hs * 81 + we]
                  - sat[he * 81 + ws] + sat[hs * 81 + ws];
        float val = win * rinv[r];
        float gm  = gmean;
        if (rempty[r]) { val = 0.0f; gm = 0.0f; }
        pooled[(size_t)r * K_ + c]      = __hip_bfloat16(val);
        pooled[(size_t)r * K_ + C_ + c] = __hip_bfloat16(gm);
    }
}

// ---------------------------------------------------------------------------
// Kernel 4: GEMM (M_PAD x 4096) x (4096 x 64) with v_wmma_f32_16x16x32_bf16,
// fused tanh + W2 reduction + softplus. 126 blocks x 128 threads (4 waves);
// wave w owns output tile (16 rois) x (hidden n = 16w..16w+15).
//
// Fragment layouts per CDNA5 ISA 7.12.2 (wave32):
//   A 16x32 bf16 : lane L holds row M=L%16; K chunks {0..7}+8*sel (V0-3)
//                  and {16..23}+8*sel (V4-7), sel = L/16.
//   B 32x16 bf16 : lane L holds column N=L%16; K chunks {0..7}+8*sel, {16..23}+8*sel (as A).
//   C/D 16x16 f32: lane L holds column N=L%16, rows {0..7}+8*sel.
// ---------------------------------------------------------------------------
__global__ void __launch_bounds__(128)
gemm_mlp_k(const __bf16* __restrict__ pooled,
           const __bf16* __restrict__ w1b,
           const float* __restrict__ b1,
           const float* __restrict__ w2,
           const float* __restrict__ b2,
           float* __restrict__ out) {
    __shared__ float red[4 * 16];

    const int mtile = blockIdx.x;            // 0..125
    const int wave  = threadIdx.x >> 5;      // 0..3 -> hidden-unit group
    const int lane  = threadIdx.x & 31;
    const int l16   = lane & 15;
    const int sel   = lane >> 4;

    const __bf16* aptr = pooled + (size_t)(mtile * 16 + l16) * K_;  // roi row
    const __bf16* bptr = w1b    + (size_t)(wave * 16 + l16) * K_;   // W1 row n

    v8f acc = {};
#pragma unroll 4
    for (int k = 0; k < K_; k += 32) {
        Frag16 a, bm;
        a.u[0]  = *(const uint4*)(aptr + k + sel * 8);        // K  0..7  (+8*sel)
        a.u[1]  = *(const uint4*)(aptr + k + 16 + sel * 8);   // K 16..23 (+8*sel)
        bm.u[0] = *(const uint4*)(bptr + k + sel * 8);        // K  0..7  (+8*sel)
        bm.u[1] = *(const uint4*)(bptr + k + 16 + sel * 8);   // K 16..23 (+8*sel)
        __builtin_prefetch(aptr + k + 128, 0, 0);
        acc = __builtin_amdgcn_wmma_f32_16x16x32_bf16(
                  false, a.v, false, bm.v, (short)0, acc, false, false);
    }

    // Epilogue: this lane holds hidden unit n = wave*16 + l16,
    // rows m = acc[i] + 8*sel of the roi tile.
    const int   n    = wave * 16 + l16;
    const float bias = b1[n];
    const float w2n  = w2[n];
    float part[8];
#pragma unroll
    for (int i = 0; i < 8; ++i) part[i] = w2n * tanhf(acc[i] + bias);

    // Reduce across the 16 lanes of each half (rows stay fixed, n varies).
#pragma unroll
    for (int off = 1; off < 16; off <<= 1) {
#pragma unroll
        for (int i = 0; i < 8; ++i) part[i] += __shfl_xor(part[i], off, 32);
    }
    if (l16 == 0) {
#pragma unroll
        for (int i = 0; i < 8; ++i) red[wave * 16 + sel * 8 + i] = part[i];
    }
    __syncthreads();

    if (threadIdx.x < 16) {
        int m = threadIdx.x;
        float s = red[m] + red[16 + m] + red[32 + m] + red[48 + m] + b2[0];
        float sp = fmaxf(s, 0.0f) + log1pf(expf(-fabsf(s)));   // softplus
        int r = mtile * 16 + m;
        if (r < R_) out[r] = sp;
    }
}

// ---------------------------------------------------------------------------
extern "C" void kernel_launch(void* const* d_in, const int* in_sizes, int n_in,
                              void* d_out, int out_size, void* d_ws, size_t ws_size,
                              hipStream_t stream) {
    const float* feat = (const float*)d_in[0];
    const float* rois = (const float*)d_in[1];
    const float* W1   = (const float*)d_in[2];
    const float* b1   = (const float*)d_in[3];
    const float* W2   = (const float*)d_in[4];
    const float* b2   = (const float*)d_in[5];
    float* out = (float*)d_out;

    // Workspace layout (~17.1 MB total)
    char* p = (char*)d_ws;
    __hip_bfloat16* pooled = (__hip_bfloat16*)p; p += (size_t)M_PAD * K_ * sizeof(__hip_bfloat16);
    __hip_bfloat16* w1b    = (__hip_bfloat16*)p; p += (size_t)NHID  * K_ * sizeof(__hip_bfloat16);
    int*   rb     = (int*)p;   p += R_ * sizeof(int);
    int*   rhs    = (int*)p;   p += R_ * sizeof(int);
    int*   rhe    = (int*)p;   p += R_ * sizeof(int);
    int*   rws    = (int*)p;   p += R_ * sizeof(int);
    int*   rwe    = (int*)p;   p += R_ * sizeof(int);
    int*   rempty = (int*)p;   p += R_ * sizeof(int);
    float* rinv   = (float*)p;

    prep_rois_k<<<(R_ + 255) / 256, 256, 0, stream>>>(
        rois, rb, rhs, rhe, rws, rwe, rempty, rinv);

    prep_weights_k<<<(NHID * K_ + 255) / 256, 256, 0, stream>>>(W1, w1b, pooled);

    plane_pool_k<<<B_ * C_, 256, 0, stream>>>(
        feat, rb, rhs, rhe, rws, rwe, rempty, rinv, pooled);

    gemm_mlp_k<<<M_PAD / 16, 128, 0, stream>>>(
        (const __bf16*)pooled, (const __bf16*)w1b, b1, W2, b2, out);
}
